// TemporalCrossAttention_36344013259300
// MI455X (gfx1250) — matrix operand order repaired and checked by reading, not compiled, not run on hardware
//
#include <hip/hip_runtime.h>
#include <hip/hip_bf16.h>
#include <math.h>

// Problem dims (match reference)
#define BB 4
#define TT 1024
#define NNq 1024
#define DDm 1024
#define LLd 512
#define HHn 16
#define HDd 64
#define TEe 1024

typedef __attribute__((ext_vector_type(8)))  float    v8f;
typedef __attribute__((ext_vector_type(16))) _Float16 v16h;
typedef __attribute__((ext_vector_type(8)))  _Float16 v8h;

__device__ __forceinline__ v16h mk16(v8h a, v8h b) {
  v16h r;
#pragma unroll
  for (int j = 0; j < 8; ++j) { r[j] = a[j]; r[j + 8] = b[j]; }
  return r;
}

// ---------------------------------------------------------------------------
// f32 -> f16 convert
// ---------------------------------------------------------------------------
__global__ void cvt_f16_kernel(const float* __restrict__ in, _Float16* __restrict__ out, int n) {
  int i = blockIdx.x * 256 + threadIdx.x;
  if (i < n) out[i] = (_Float16)in[i];
}

// ---------------------------------------------------------------------------
// LayerNorm (block per row) -> f16
// ---------------------------------------------------------------------------
__global__ __launch_bounds__(256)
void ln_f16_kernel(const float* __restrict__ x, const float* __restrict__ g,
                   const float* __restrict__ b, _Float16* __restrict__ out, int cols) {
  int row = blockIdx.x;
  const float* xr = x + (size_t)row * cols;
  float s = 0.f, s2 = 0.f;
  for (int c = threadIdx.x; c < cols; c += 256) { float v = xr[c]; s += v; s2 += v * v; }
  __shared__ float red0[256];
  __shared__ float red1[256];
  red0[threadIdx.x] = s; red1[threadIdx.x] = s2;
  __syncthreads();
  for (int off = 128; off > 0; off >>= 1) {
    if ((int)threadIdx.x < off) {
      red0[threadIdx.x] += red0[threadIdx.x + off];
      red1[threadIdx.x] += red1[threadIdx.x + off];
    }
    __syncthreads();
  }
  float mean = red0[0] / (float)cols;
  float var  = red1[0] / (float)cols - mean * mean;
  float rstd = rsqrtf(var + 1e-5f);
  for (int c = threadIdx.x; c < cols; c += 256) {
    float v = (xr[c] - mean) * rstd * g[c] + b[c];
    out[(size_t)row * cols + c] = (_Float16)v;
  }
}

// ---------------------------------------------------------------------------
// AdaLN (LN * (1+scale) + shift) then SiLU -> f16
// ---------------------------------------------------------------------------
__global__ __launch_bounds__(256)
void adaln_silu_kernel(const float* __restrict__ y, const float* __restrict__ g,
                       const float* __restrict__ b, const float* __restrict__ embo,
                       _Float16* __restrict__ out, int cols, int Trows) {
  int row = blockIdx.x;
  int bidx = row / Trows;
  const float* yr = y + (size_t)row * cols;
  float s = 0.f, s2 = 0.f;
  for (int c = threadIdx.x; c < cols; c += 256) { float v = yr[c]; s += v; s2 += v * v; }
  __shared__ float red0[256];
  __shared__ float red1[256];
  red0[threadIdx.x] = s; red1[threadIdx.x] = s2;
  __syncthreads();
  for (int off = 128; off > 0; off >>= 1) {
    if ((int)threadIdx.x < off) {
      red0[threadIdx.x] += red0[threadIdx.x + off];
      red1[threadIdx.x] += red1[threadIdx.x + off];
    }
    __syncthreads();
  }
  float mean = red0[0] / (float)cols;
  float var  = red1[0] / (float)cols - mean * mean;
  float rstd = rsqrtf(var + 1e-5f);
  const float* sc = embo + (size_t)bidx * 2 * cols;
  const float* sh = sc + cols;
  for (int c = threadIdx.x; c < cols; c += 256) {
    float v = (yr[c] - mean) * rstd * g[c] + b[c];
    v = v * (1.0f + sc[c]) + sh[c];
    float sil = v / (1.0f + __expf(-v));
    out[(size_t)row * cols + c] = (_Float16)sil;
  }
}

// ---------------------------------------------------------------------------
// SiLU elementwise (f32 -> f32), for emb
// ---------------------------------------------------------------------------
__global__ void silu_kernel(const float* __restrict__ in, float* __restrict__ out, int n) {
  int i = blockIdx.x * 256 + threadIdx.x;
  if (i < n) { float v = in[i]; out[i] = v / (1.0f + __expf(-v)); }
}

// ---------------------------------------------------------------------------
// Tiny GEMV for emb path: out[b][j] = sum_i semb[b][i]*W[i][j] + bias[j]
// ---------------------------------------------------------------------------
__global__ void emb_gemv_kernel(const float* __restrict__ semb, const float* __restrict__ W,
                                const float* __restrict__ bias, float* __restrict__ out,
                                int K, int Nc) {
  int j = blockIdx.x * 256 + threadIdx.x;   // over BB * Nc
  int bb = j / Nc;
  int col = j - bb * Nc;
  const float* e = semb + (size_t)bb * K;
  float acc = bias[col];
  for (int i = 0; i < K; ++i) acc += e[i] * W[(size_t)i * Nc + col];
  out[j] = acc;
}

// ---------------------------------------------------------------------------
// WMMA GEMM: out[M x Nc] = A_f16[M x K] @ W_f16[K x Nc] + bias (+ residual)
// Block tile 128x64, 8 waves (4 M x 2 N), wave tile 32x32 = 2x2 WMMA tiles.
// ---------------------------------------------------------------------------
template <bool F16OUT, bool RESID>
__global__ __launch_bounds__(256)
void gemm_wmma(const _Float16* __restrict__ A, const _Float16* __restrict__ W,
               const float* __restrict__ bias, const float* __restrict__ resid,
               void* __restrict__ outp, int M, int K, int Nc) {
  __shared__ __align__(16) _Float16 WT[64][40];   // [n][k], padded stride
  const int tid  = threadIdx.x;
  const int wid  = tid >> 5;
  const int lane = tid & 31;
  const int ln16 = lane & 15;
  const int hi   = lane >> 4;
  const int mw   = wid & 3;   // 0..3
  const int nw   = wid >> 2;  // 0..1
  const int mblk = blockIdx.x * 128;
  const int nblk = blockIdx.y * 64;

  v8f acc[2][2] = {};

  for (int kk = 0; kk < K; kk += 32) {
    // cooperative load of W[kk..kk+31][nblk..nblk+63], transposed into LDS
    {
      int kloc = tid >> 3;          // 0..31
      int n0   = (tid & 7) * 8;     // 0..56
      v8h w8 = *(const v8h*)(W + (size_t)(kk + kloc) * Nc + nblk + n0);
#pragma unroll
      for (int j = 0; j < 8; ++j) WT[n0 + j][kloc] = w8[j];
    }
    __syncthreads();

    // A fragments straight from global in 16-bit A-matrix layout
    v16h afrag[2];
#pragma unroll
    for (int mi = 0; mi < 2; ++mi) {
      int mrow = mblk + mw * 32 + mi * 16 + ln16;
      const _Float16* ap = A + (size_t)mrow * K + kk;
      v8h lo  = *(const v8h*)(ap + 16 * hi);
      v8h hi8 = *(const v8h*)(ap + 16 * hi + 8);
      afrag[mi] = mk16(lo, hi8);
    }
    // B fragments from transposed LDS tile (contiguous per lane)
    v16h bfrag[2];
#pragma unroll
    for (int ni = 0; ni < 2; ++ni) {
      int ncol = nw * 32 + ni * 16 + ln16;
      const _Float16* bp = &WT[ncol][16 * hi];
      v8h lo  = *(const v8h*)(bp);
      v8h hi8 = *(const v8h*)(bp + 8);
      bfrag[ni] = mk16(lo, hi8);
    }
#pragma unroll
    for (int mi = 0; mi < 2; ++mi)
#pragma unroll
      for (int ni = 0; ni < 2; ++ni)
        acc[mi][ni] = __builtin_amdgcn_wmma_f32_16x16x32_f16(
            false, afrag[mi], false, bfrag[ni], (short)0, acc[mi][ni], false, false);
    __syncthreads();
  }

  // epilogue
#pragma unroll
  for (int mi = 0; mi < 2; ++mi) {
#pragma unroll
    for (int ni = 0; ni < 2; ++ni) {
      int ncol = nblk + nw * 32 + ni * 16 + ln16;
      float bv = bias[ncol];
#pragma unroll
      for (int r = 0; r < 8; ++r) {
        int mrow = mblk + mw * 32 + mi * 16 + r + 8 * hi;
        float v = acc[mi][ni][r] + bv;
        if (F16OUT) {
          ((_Float16*)outp)[(size_t)mrow * Nc + ncol] = (_Float16)v;
        } else {
          if (RESID) v += resid[(size_t)mrow * Nc + ncol];
          ((float*)outp)[(size_t)mrow * Nc + ncol] = v;
        }
      }
    }
  }
}

// ---------------------------------------------------------------------------
// Fused flash attention: block = (b,h, 128 query rows), 8 waves, each wave
// owns 16 rows x full HD=64. Online softmax in C-matrix register layout.
// ---------------------------------------------------------------------------
__global__ __launch_bounds__(256)
void attn_kernel(const _Float16* __restrict__ q, const _Float16* __restrict__ k,
                 const _Float16* __restrict__ v, const float* __restrict__ mask,
                 float* __restrict__ y) {
  __shared__ __align__(16) _Float16 VT[64][40];       // V^T tile: [hd][n_local]
  __shared__ __align__(16) _Float16 Psh[8][16][40];   // per-wave P slab [m][n_local]

  const int tid  = threadIdx.x;
  const int wid  = tid >> 5;
  const int lane = tid & 31;
  const int ln16 = lane & 15;
  const int hi   = lane >> 4;
  const int bh = blockIdx.x;
  const int b  = bh >> 4;     // /HHn
  const int h  = bh & 15;
  const int tbase = blockIdx.y * 128 + wid * 16;

  // Q fragments for this wave's 16 rows, K-dim = HD = 64 (2 WMMA k-steps)
  v16h qa[2];
  {
    const _Float16* qp = q + ((size_t)(b * TT + tbase + ln16)) * DDm + h * HDd;
#pragma unroll
    for (int kkI = 0; kkI < 2; ++kkI) {
      v8h lo  = *(const v8h*)(qp + kkI * 32 + 16 * hi);
      v8h hi8 = *(const v8h*)(qp + kkI * 32 + 16 * hi + 8);
      qa[kkI] = mk16(lo, hi8);
    }
  }

  v8f acc[4] = {};
  float mrow[8], lrow[8];
#pragma unroll
  for (int r = 0; r < 8; ++r) { mrow[r] = -3.0e38f; lrow[r] = 0.f; }

  for (int nb = 0; nb < NNq; nb += 32) {
    __syncthreads();   // previous-iteration VT reads done
    // stage V^T tile: V[nb..nb+31][h*64..+63] -> VT[hd][n_local]
    {
      int nl = tid >> 3;
      int d0 = (tid & 7) * 8;
      v8h vv = *(const v8h*)(v + ((size_t)(b * NNq + nb + nl)) * DDm + h * HDd + d0);
#pragma unroll
      for (int j = 0; j < 8; ++j) VT[d0 + j][nl] = vv[j];
    }
    __syncthreads();

    // S = Q . K^T for two 16-col tiles (K read directly in B-layout)
    v8f S[2];
#pragma unroll
    for (int jt = 0; jt < 2; ++jt) {
      int nrow = nb + jt * 16 + ln16;
      const _Float16* kp = k + ((size_t)(b * NNq + nrow)) * DDm + h * HDd + 16 * hi;
      v16h kb0 = mk16(*(const v8h*)(kp), *(const v8h*)(kp + 8));
      v16h kb1 = mk16(*(const v8h*)(kp + 32), *(const v8h*)(kp + 40));
      v8f s = {};
      s = __builtin_amdgcn_wmma_f32_16x16x32_f16(false, qa[0], false, kb0, (short)0, s, false, false);
      s = __builtin_amdgcn_wmma_f32_16x16x32_f16(false, qa[1], false, kb1, (short)0, s, false, false);
      S[jt] = s;
    }

    // online softmax across the 32 new columns, in C-layout registers
#pragma unroll
    for (int r = 0; r < 8; ++r) {
      int row = r + 8 * hi;
      const float* mp = mask + ((size_t)(b * TT + tbase + row)) * NNq + nb;
      float s0 = S[0][r] * 0.125f + (1.0f - mp[ln16])      * -100000.0f;
      float s1 = S[1][r] * 0.125f + (1.0f - mp[16 + ln16]) * -100000.0f;
      float mx = fmaxf(s0, s1);
#pragma unroll
      for (int o = 8; o > 0; o >>= 1) mx = fmaxf(mx, __shfl_xor(mx, o, 32));
      float mnew = fmaxf(mrow[r], mx);
      float corr = __expf(mrow[r] - mnew);
      float p0 = __expf(s0 - mnew);
      float p1 = __expf(s1 - mnew);
      float ssum = p0 + p1;
#pragma unroll
      for (int o = 8; o > 0; o >>= 1) ssum += __shfl_xor(ssum, o, 32);
      lrow[r] = lrow[r] * corr + ssum;
      mrow[r] = mnew;
#pragma unroll
      for (int c = 0; c < 4; ++c) acc[c][r] *= corr;
      Psh[wid][row][ln16]      = (_Float16)p0;
      Psh[wid][row][16 + ln16] = (_Float16)p1;
    }
    // per-wave LDS store->load ordering (cross-lane within wave)
    asm volatile("s_wait_dscnt 0x0" ::: "memory");

    // P (A-layout from LDS) x V^T (B-layout from LDS), K-dim = 32
    v16h pa;
    {
      const _Float16* pp = &Psh[wid][ln16][0];
      pa = mk16(*(const v8h*)(pp + 16 * hi), *(const v8h*)(pp + 16 * hi + 8));
    }
#pragma unroll
    for (int c = 0; c < 4; ++c) {
      const _Float16* vp = &VT[c * 16 + ln16][16 * hi];
      v16h vb = mk16(*(const v8h*)(vp), *(const v8h*)(vp + 8));
      acc[c] = __builtin_amdgcn_wmma_f32_16x16x32_f16(false, pa, false, vb, (short)0, acc[c], false, false);
    }
  }

  // finalize: divide by row sums, write y (f32)
#pragma unroll
  for (int r = 0; r < 8; ++r) {
    float inv = 1.0f / lrow[r];
    int trow = tbase + r + 8 * hi;
    float* yp = y + ((size_t)(b * TT + trow)) * DDm + h * HDd;
#pragma unroll
    for (int c = 0; c < 4; ++c) yp[c * 16 + ln16] = acc[c][r] * inv;
  }
}

// ---------------------------------------------------------------------------
// Launcher
// ---------------------------------------------------------------------------
extern "C" void kernel_launch(void* const* d_in, const int* in_sizes, int n_in,
                              void* d_out, int out_size, void* d_ws, size_t ws_size,
                              hipStream_t stream) {
  const float* x      = (const float*)d_in[0];
  const float* xf     = (const float*)d_in[1];
  const float* emb    = (const float*)d_in[2];
  const float* smask  = (const float*)d_in[3];
  const float* ln_g   = (const float*)d_in[4];
  const float* ln_b   = (const float*)d_in[5];
  const float* aln_g  = (const float*)d_in[6];
  const float* aln_b  = (const float*)d_in[7];
  const float* n2_g   = (const float*)d_in[8];
  const float* n2_b   = (const float*)d_in[9];
  const float* Wq     = (const float*)d_in[10];
  const float* bq     = (const float*)d_in[11];
  const float* Wk     = (const float*)d_in[12];
  const float* bk     = (const float*)d_in[13];
  const float* Wv     = (const float*)d_in[14];
  const float* bv     = (const float*)d_in[15];
  const float* emb_W  = (const float*)d_in[16];
  const float* emb_b  = (const float*)d_in[17];
  const float* out_W  = (const float*)d_in[18];
  const float* out_b  = (const float*)d_in[19];

  const size_t BT = (size_t)BB * TT;   // 4096
  const size_t BN = (size_t)BB * NNq;  // 4096

  // workspace carving
  char* ws = (char*)d_ws;
  size_t off = 0;
  auto carve = [&](size_t bytes) -> char* {
    char* p = ws + off;
    off = (off + bytes + 255) & ~(size_t)255;
    return p;
  };
  _Float16* xn16  = (_Float16*)carve(BT * DDm * 2);
  _Float16* xfn16 = (_Float16*)carve(BN * LLd * 2);
  _Float16* wq16  = (_Float16*)carve((size_t)DDm * DDm * 2);
  _Float16* wk16  = (_Float16*)carve((size_t)LLd * DDm * 2);
  _Float16* wv16  = (_Float16*)carve((size_t)LLd * DDm * 2);
  _Float16* wo16  = (_Float16*)carve((size_t)DDm * DDm * 2);
  _Float16* q16   = (_Float16*)carve(BT * DDm * 2);
  _Float16* k16   = (_Float16*)carve(BN * DDm * 2);
  _Float16* v16   = (_Float16*)carve(BN * DDm * 2);
  float*    y32   = (float*)   carve(BT * DDm * 4);
  float*    semb  = (float*)   carve((size_t)BB * TEe * 4);
  float*    embo  = (float*)   carve((size_t)BB * 2 * DDm * 4);
  _Float16* hs16  = (_Float16*)carve(BT * DDm * 2);
  (void)ws_size; (void)n_in; (void)in_sizes; (void)out_size;

  // 1) weight conversions
  {
    int n = DDm * DDm;
    cvt_f16_kernel<<<(n + 255) / 256, 256, 0, stream>>>(Wq, wq16, n);
    cvt_f16_kernel<<<(n + 255) / 256, 256, 0, stream>>>(out_W, wo16, n);
    int m = LLd * DDm;
    cvt_f16_kernel<<<(m + 255) / 256, 256, 0, stream>>>(Wk, wk16, m);
    cvt_f16_kernel<<<(m + 255) / 256, 256, 0, stream>>>(Wv, wv16, m);
  }
  // 2) LayerNorms -> f16
  ln_f16_kernel<<<(int)BT, 256, 0, stream>>>(x, ln_g, ln_b, xn16, DDm);
  ln_f16_kernel<<<(int)BN, 256, 0, stream>>>(xf, aln_g, aln_b, xfn16, LLd);

  // 3) Q/K/V projections (WMMA)
  dim3 gq((int)BT / 128, DDm / 64);
  gemm_wmma<true, false><<<gq, 256, 0, stream>>>(xn16, wq16, bq, nullptr, q16, (int)BT, DDm, DDm);
  gemm_wmma<true, false><<<gq, 256, 0, stream>>>(xfn16, wk16, bk, nullptr, k16, (int)BN, LLd, DDm);
  gemm_wmma<true, false><<<gq, 256, 0, stream>>>(xfn16, wv16, bv, nullptr, v16, (int)BN, LLd, DDm);

  // 4) emb path: silu + GEMV
  {
    int n = BB * TEe;
    silu_kernel<<<(n + 255) / 256, 256, 0, stream>>>(emb, semb, n);
    int m = BB * 2 * DDm;
    emb_gemv_kernel<<<(m + 255) / 256, 256, 0, stream>>>(semb, emb_W, emb_b, embo, TEe, 2 * DDm);
  }

  // 5) fused flash attention -> y (f32)
  {
    dim3 ga(BB * HHn, TT / 128);
    attn_kernel<<<ga, 256, 0, stream>>>(q16, k16, v16, smask, y32);
  }

  // 6) AdaLN + SiLU -> f16
  adaln_silu_kernel<<<(int)BT, 256, 0, stream>>>(y32, n2_g, n2_b, embo, hs16, DDm, TT);

  // 7) output projection + bias + residual -> d_out (f32)
  gemm_wmma<false, true><<<gq, 256, 0, stream>>>(hs16, wo16, out_b, x, (float*)d_out, (int)BT, DDm, DDm);
}
